// AdditiveAttention_45681272160514
// MI455X (gfx1250) — matrix operand order repaired and checked by reading, not compiled, not run on hardware
//
#include <hip/hip_runtime.h>
#include <hip/hip_bf16.h>

// Shapes from the reference
#define BQ  4
#define QL  1024
#define KL  1024
#define DQK 256
#define HH  64
#define DVV 256

typedef __attribute__((ext_vector_type(16))) __bf16 v16bf;
typedef __attribute__((ext_vector_type(8)))  __bf16 v8bf;
typedef __attribute__((ext_vector_type(8)))  float  v8f;

__device__ __forceinline__ float fast_tanh(float x) {
#if __has_builtin(__builtin_amdgcn_tanhf)
    return __builtin_amdgcn_tanhf(x);   // v_tanh_f32 (gfx1250 TRANS op)
#else
    float e = __expf(-2.0f * fabsf(x));
    float t = (1.0f - e) / (1.0f + e);
    return copysignf(t, x);
#endif
}

__device__ __forceinline__ v16bf make_v16(v8bf a, v8bf b) {
    return __builtin_shufflevector(a, b, 0,1,2,3,4,5,6,7,8,9,10,11,12,13,14,15);
}

// A-fragment (16x32 bf16, row-major LDS tile [16][stride]).
// Lane L<16: M=L, elems 0-7 = K=ks..ks+7, elems 8-15 = K=ks+16..ks+23.
// Lane L>=16: M=L-16, elems 0-7 = K=ks+8..15, elems 8-15 = K=ks+24..31.
__device__ __forceinline__ v16bf load_afrag(const __bf16* lds, int stride, int lane, int ks) {
    int M  = lane & 15;
    int k1 = ks + ((lane >> 4) & 1) * 8;
    const __bf16* p = lds + M * stride + k1;
    v8bf c0 = *(const v8bf*)(p);
    v8bf c1 = *(const v8bf*)(p + 16);
    return make_v16(c0, c1);
}

// B-fragment (32x16 bf16) from K-contiguous transposed LDS tile [n][stride].
// Lane L<16: N=L, elems 0-7 = K=ks..ks+7, elems 8-15 = K=ks+16..ks+23.  Lane L>=16: N=L-16, elems 0-7 = K=ks+8..15, elems 8-15 = K=ks+24..31.
__device__ __forceinline__ v16bf load_bfrag(const __bf16* ldsT, int stride, int lane, int nb, int ks) {
    int n  = nb + (lane & 15);
    int kb = ks + ((lane >> 4) & 1) * 8;
    const __bf16* p = ldsT + n * stride + kb;
    v8bf c0 = *(const v8bf*)(p);
    v8bf c1 = *(const v8bf*)(p + 16);
    return make_v16(c0, c1);
}

// ---------------------------------------------------------------------------
// Kernel 1: fused projections. Rows 0..4095 -> qp = queries@W_q,
// rows 4096..8191 -> kp = keys@W_k. Output f32 [8192][64] in workspace.
// 4 waves/block, each wave owns one 16-col N-tile of H=64.
// ---------------------------------------------------------------------------
__global__ __launch_bounds__(128) void proj_kernel(
    const float* __restrict__ queries, const float* __restrict__ keys,
    const float* __restrict__ Wq, const float* __restrict__ Wk,
    float* __restrict__ proj) {
    __shared__ __bf16 xtile[16 * DQK];     // 8 KB, row-major
    __shared__ __bf16 wtileT[HH * DQK];    // 32 KB, [n][k] K-contiguous

    int t  = threadIdx.x;
    int rb = blockIdx.x * 16;
    bool isq = rb < BQ * QL;
    const float* X = isq ? (queries + (size_t)rb * DQK)
                         : (keys + (size_t)(rb - BQ * QL) * DQK);
    const float* W = isq ? Wq : Wk;

    for (int idx = t; idx < 16 * DQK; idx += 128)
        xtile[idx] = (__bf16)X[idx];                       // rows contiguous
    for (int idx = t; idx < HH * DQK; idx += 128) {
        int k = idx >> 6, n = idx & 63;                    // W is [256][64]
        wtileT[n * DQK + k] = (__bf16)W[idx];
    }
    __syncthreads();

    int wave = t >> 5, lane = t & 31;
    int nb = wave * 16;
    v8f acc = {};
    #pragma unroll
    for (int ks = 0; ks < DQK; ks += 32) {
        v16bf a = load_afrag(xtile, DQK, lane, ks);
        v16bf b = load_bfrag(wtileT, DQK, lane, nb, ks);
        acc = __builtin_amdgcn_wmma_f32_16x16x32_bf16(false, a, false, b,
                                                      (short)0, acc, false, false);
    }
    int col = nb + (lane & 15);
    int rhi = ((lane >> 4) & 1) * 8;       // C layout: lanes 16-31 hold M=v+8
    #pragma unroll
    for (int i = 0; i < 8; ++i)
        proj[(size_t)(rb + rhi + i) * HH + col] = acc[i];
}

// ---------------------------------------------------------------------------
// Kernel 2: fused scores + online softmax + attn@V (flash-style).
// Block = (qtile, batch); 8 waves; each wave owns 32 output columns of DV.
// ---------------------------------------------------------------------------
__global__ __launch_bounds__(256) void attn_kernel(
    const float* __restrict__ proj, const float* __restrict__ values,
    const float* __restrict__ wv, float* __restrict__ out) {
    __shared__ float  qt[16 * HH];         // 4 KB
    __shared__ float  ktile[64 * 65];      // 16.25 KB, padded: conflict-free col reads
    __shared__ float  sc[16 * 64];         // 4 KB scores
    __shared__ __bf16 ptile[16 * 64];      // 2 KB probs (WMMA A)
    __shared__ __bf16 vtileT[DVV * 72];    // 36 KB, [n][k], stride 72 => 16B-aligned rows
    __shared__ float  wv_s[HH];
    __shared__ float  m_s[16], l_s[16], rs_s[16], linv_s[16];

    int t  = threadIdx.x;
    int b  = blockIdx.y;
    int q0 = blockIdx.x * 16;
    const float* qp = proj + (size_t)(b * QL + q0) * HH;
    const float* kp = proj + (size_t)(BQ * QL + b * KL) * HH;
    const float* V  = values + (size_t)b * KL * DVV;

    for (int idx = t; idx < 16 * HH; idx += 256) qt[idx] = qp[idx];
    if (t < HH) wv_s[t] = wv[t];
    if (t < 16) { m_s[t] = -3.0e38f; l_s[t] = 0.0f; }

    int wave = t >> 5, lane = t & 31;
    int nb0 = wave * 32;
    int rhi = ((lane >> 4) & 1) * 8;
    v8f acc0 = {}, acc1 = {};

    for (int kt = 0; kt < KL; kt += 64) {
        __syncthreads();   // also protects qt init / prior-iter fragment reads
        for (int idx = t; idx < 64 * HH; idx += 256) {          // kp tile
            int c = idx >> 6, h = idx & 63;
            ktile[c * 65 + h] = kp[(size_t)(kt + c) * HH + h];
        }
        for (int idx = t; idx < 64 * DVV; idx += 256) {         // V tile (transposed bf16)
            int k = idx >> 8, n = idx & 255;
            vtileT[n * 72 + k] = (__bf16)V[(size_t)(kt + k) * DVV + n];
        }
        __syncthreads();

        // scores: 16x64 tile, 4 per thread; hot loop: fma + v_tanh_f32
        #pragma unroll
        for (int j = 0; j < 4; ++j) {
            int id = t + 256 * j;
            int r = id >> 6, c = id & 63;
            const float* qr = qt + r * HH;
            const float* kc = ktile + c * 65;
            float s = 0.0f;
            #pragma unroll 8
            for (int h = 0; h < HH; ++h)
                s += wv_s[h] * fast_tanh(qr[h] + kc[h]);
            sc[r * 64 + c] = s;
        }
        __syncthreads();

        // online softmax per row
        if (t < 16) {
            float mo = m_s[t], mn = mo;
            for (int c = 0; c < 64; ++c) mn = fmaxf(mn, sc[t * 64 + c]);
            float scale = __expf(mo - mn);
            float l = l_s[t] * scale;
            for (int c = 0; c < 64; ++c) {
                float p = __expf(sc[t * 64 + c] - mn);
                l += p;
                ptile[t * 64 + c] = (__bf16)p;
            }
            m_s[t] = mn; l_s[t] = l; rs_s[t] = scale;
        }
        __syncthreads();

        // rescale running accumulators (C layout: elem i -> row i + rhi)
        #pragma unroll
        for (int i = 0; i < 8; ++i) {
            float f = rs_s[rhi + i];
            acc0[i] *= f; acc1[i] *= f;
        }
        // p(16x64) @ V(64x32 per wave): 4 WMMAs per tile per wave
        #pragma unroll
        for (int ks = 0; ks < 64; ks += 32) {
            v16bf a  = load_afrag(ptile, 64, lane, ks);
            v16bf b0 = load_bfrag(vtileT, 72, lane, nb0, ks);
            acc0 = __builtin_amdgcn_wmma_f32_16x16x32_bf16(false, a, false, b0,
                                                           (short)0, acc0, false, false);
            v16bf b1 = load_bfrag(vtileT, 72, lane, nb0 + 16, ks);
            acc1 = __builtin_amdgcn_wmma_f32_16x16x32_bf16(false, a, false, b1,
                                                           (short)0, acc1, false, false);
        }
    }
    __syncthreads();
    if (t < 16) linv_s[t] = 1.0f / l_s[t];
    __syncthreads();

    int col = lane & 15;
    float* orow = out + (size_t)(b * QL + q0) * DVV;
    #pragma unroll
    for (int i = 0; i < 8; ++i) {
        float inv = linv_s[rhi + i];
        orow[(size_t)(rhi + i) * DVV + nb0 + col]      = acc0[i] * inv;
        orow[(size_t)(rhi + i) * DVV + nb0 + 16 + col] = acc1[i] * inv;
    }
}

extern "C" void kernel_launch(void* const* d_in, const int* in_sizes, int n_in,
                              void* d_out, int out_size, void* d_ws, size_t ws_size,
                              hipStream_t stream) {
    const float* queries = (const float*)d_in[0];   // [4,1024,256]
    const float* keys    = (const float*)d_in[1];   // [4,1024,256]
    const float* values  = (const float*)d_in[2];   // [4,1024,256]
    const float* Wq      = (const float*)d_in[3];   // [256,64]
    const float* Wk      = (const float*)d_in[4];   // [256,64]
    const float* wv      = (const float*)d_in[5];   // [64]
    float* out  = (float*)d_out;                    // [4,1024,256] f32
    float* proj = (float*)d_ws;                     // needs 2*4096*64*4 = 2 MB scratch

    proj_kernel<<<(2 * BQ * QL) / 16, 128, 0, stream>>>(queries, keys, Wq, Wk, proj);
    attn_kernel<<<dim3(QL / 16, BQ), 256, 0, stream>>>(proj, values, wv, out);
}
